// MPN_6734508720280
// MI455X (gfx1250) — matrix operand order repaired and checked by reading, not compiled, not run on hardware
//
#include <hip/hip_runtime.h>
#include <hip/hip_bf16.h>

// ---------------------------------------------------------------------------
// MPN (message-passing GNN) for MI455X / gfx1250, wave32 + WMMA f16->f32.
//
// Strategy (bandwidth-limited problem, ~156 GFLOP total vs 23.3 TB/s HBM):
//  * BatchNorm uses exact batch stats -> device-wide reduction between every
//    linear layer -> multi-pass pipeline with recompute (inputs are smaller
//    than intermediates, and M (1.2MB) + weights live in the 192MB L2).
//  * bias before BN cancels (BN(x+b)==BN(x)) -> l1/l2 biases dropped.
//  * Each wave processes 16-edge tiles; GEMMs are v_wmma_f32_16x16x32_f16
//    with K zero-padded to 32 and HID=20 split over two 16-wide N tiles.
//  * Activation relayout D-frag -> A-frag goes through a per-wave 16x32 f16
//    LDS tile (cross-lane transpose), guarded by s_wait_dscnt.
//  * Scatter-add (segment_sum) via global_atomic_add_f32 (L2-resident table).
//
// Flat param layout assumes JAX pytree flattening (sorted dict keys):
//   be1, be2, g1, g2, l1.W, l1.b, l2.W, l2.b, l3.W, l3.b
// ---------------------------------------------------------------------------

#define N_NODES_C   100000
#define N_EDGES_C   3200000
#define NUM_LAYER_C 20
#define NTILES_C    (N_EDGES_C / 16)   // 200000, E is an exact multiple of 16

typedef __attribute__((ext_vector_type(16))) _Float16 v16h;
typedef __attribute__((ext_vector_type(8)))  float    v8f;
typedef _Float16 f16_t;

// params_e (fan_in 9): offsets in floats
#define PE_BE1 0
#define PE_BE2 20
#define PE_G1  40
#define PE_G2  60
#define PE_W1  80
#define PE_W2  280
#define PE_W3  700
#define PE_B3  760
// params_v (fan_in 6)
#define PV_BE1 0
#define PV_BE2 20
#define PV_G1  40
#define PV_G2  60
#define PV_W1  80
#define PV_W2  220
#define PV_W3  640
#define PV_B3  700
// params_out (3->3->2)
#define PO_BE1 0
#define PO_BE2 3
#define PO_G1  6
#define PO_G2  9
#define PO_W1  12
#define PO_W2  24
#define PO_W3  36
#define PO_B3  42

__device__ inline v8f wmma32(const v16h& a, const v16h& b, const v8f& c) {
  // D = A(16x32 f16) * B(32x16 f16) + C(16x16 f32)
  return __builtin_amdgcn_wmma_f32_16x16x32_f16(false, a, false, b, (short)0, c,
                                                false, false);
}

__device__ inline v16h vz16() {
  v16h a;
#pragma unroll
  for (int i = 0; i < 16; ++i) a[i] = (f16_t)0.f;
  return a;
}
__device__ inline v8f vz8() {
  v8f d;
#pragma unroll
  for (int i = 0; i < 8; ++i) d[i] = 0.f;
  return d;
}

// B fragment from fp32 weight W[fan_in][fan_out] (row-major), zero-padded to
// K=32, N-tile `ntile` selects output channels [16*ntile, 16*ntile+16).
// Assumed dense-B layout (per ISA SWMMAC-B pattern scaled to K=32):
// lane: col = lane%16, K = (element/8)*16 + (lane/16)*8 + element%8.
__device__ inline v16h make_bfrag(const float* __restrict__ W, int fan_in,
                                  int fan_out, int ntile, int lane) {
  const int n = (lane & 15) + ntile * 16;
  const int kbase = (lane >> 4) * 8;
  v16h b;
#pragma unroll
  for (int e = 0; e < 16; ++e) {
    const int k = (e >> 3) * 16 + kbase + (e & 7);
    const float w = (k < fan_in && n < fan_out) ? W[k * fan_out + n] : 0.f;
    b[e] = (f16_t)w;
  }
  return b;
}

// A fragment from a per-wave 16x32 f16 LDS tile (row-major).
// ISA 16-bit A layout: lane row = lane%16; element e -> K = (e/8)*16 + (lane/16)*8 + e%8.
__device__ inline v16h afrag_lds(const f16_t* __restrict__ t, int lane) {
  const int row = lane & 15;
  const int half = lane >> 4;
  v16h a;
#pragma unroll
  for (int e = 0; e < 16; ++e) {
    const int k = (e >> 3) * 16 + half * 8 + (e & 7);
    a[e] = t[row * 32 + k];
  }
  return a;
}

// A fragment for cat[Mi, Mj, H] (K=9, rest zero). half0 covers K=0..7, half1 K=8.
__device__ inline v16h afrag_edge(const float* __restrict__ M,
                                  const float* __restrict__ H,
                                  const int* __restrict__ src,
                                  const int* __restrict__ dst, int e0, int half) {
  v16h a = vz16();
  if (half == 0) {
    const int di = dst[e0] * 3, si = src[e0] * 3;
    a[0] = (f16_t)M[di];     a[1] = (f16_t)M[di + 1]; a[2] = (f16_t)M[di + 2];
    a[3] = (f16_t)M[si];     a[4] = (f16_t)M[si + 1]; a[5] = (f16_t)M[si + 2];
    a[6] = (f16_t)H[e0 * 3]; a[7] = (f16_t)H[e0 * 3 + 1];
  } else {
    a[0] = (f16_t)H[e0 * 3 + 2];  // K = 8
  }
  return a;
}

// A fragment for cat[Mi, Hn] (K=6, rest zero).
__device__ inline v16h afrag_mihn(const float* __restrict__ M,
                                  const float* __restrict__ Hn,
                                  const int* __restrict__ dst, int e0, int half) {
  v16h a = vz16();
  if (half == 0) {
    const int di = dst[e0] * 3;
    a[0] = (f16_t)M[di];       a[1] = (f16_t)M[di + 1];     a[2] = (f16_t)M[di + 2];
    a[3] = (f16_t)Hn[e0 * 3];  a[4] = (f16_t)Hn[e0 * 3 + 1]; a[5] = (f16_t)Hn[e0 * 3 + 2];
  }
  return a;
}

// Per-lane BN scale/shift for its two N-tile channels (stats are raw sums of
// the *bias-free* pre-activation; bias cancels inside BN).
__device__ inline void bn_coeffs(const float* __restrict__ st,
                                 const float* __restrict__ g,
                                 const float* __restrict__ be, int nch, int lane,
                                 float invE, float sc[2], float sh[2]) {
#pragma unroll
  for (int nt = 0; nt < 2; ++nt) {
    const int c = (lane & 15) + nt * 16;
    float s = 0.f, h = 0.f;
    if (c < nch) {
      const float mu = st[c] * invE;
      const float var = st[32 + c] * invE - mu * mu;
      const float rs = rsqrtf(var + 1e-5f);
      s = g[c] * rs;
      h = be[c] - mu * s;
    }
    sc[nt] = s;
    sh[nt] = h;
  }
}

// ReLU(BN(D)) -> f16 LDS tile (writes all 32 columns; padded channels get 0).
__device__ inline void stage_act(f16_t* __restrict__ t, const v8f d[2],
                                 const float sc[2], const float sh[2], int lane) {
  const int half = lane >> 4;
  const int c0 = lane & 15;
#pragma unroll
  for (int nt = 0; nt < 2; ++nt) {
    const int c = c0 + nt * 16;
#pragma unroll
    for (int r = 0; r < 8; ++r) {
      float v = d[nt][r] * sc[nt] + sh[nt];
      v = fmaxf(v, 0.f);
      t[(r + half * 8) * 32 + c] = (f16_t)v;
    }
  }
}

__device__ inline void acc_stats(const v8f& d, float& s, float& q) {
#pragma unroll
  for (int r = 0; r < 8; ++r) {
    const float v = d[r];
    s += v;
    q += v * v;
  }
}

__device__ inline void flush_stats(float* __restrict__ st, const float aS[2],
                                   const float aQ[2], int lane, int nch) {
#pragma unroll
  for (int nt = 0; nt < 2; ++nt) {
    float s = aS[nt] + __shfl_xor(aS[nt], 16, 32);  // combine lane halves
    float q = aQ[nt] + __shfl_xor(aQ[nt], 16, 32);
    const int c = (lane & 15) + nt * 16;
    if (lane < 16 && c < nch) {
      atomicAdd(&st[c], s);
      atomicAdd(&st[32 + c], q);
    }
  }
}

struct LayerPtrs {
  const float* Min;
  const float* Hin;
  const int* src;
  const int* dst;
  float* Hout;
  float* Mout;
  const float* pe;
  const float* pv;
  float* st;  // [0..63]=E1{sum32,sq32} [64..127]=E2 [128..191]=V1 [192..255]=V2
};

// STAGE 0: X1 stats | 1: X2 stats | 2: write Hn + Y1 stats | 3: Y2 stats
// STAGE 4: messages + scatter-add into Mout
template <int STAGE>
__global__ __launch_bounds__(256) void mpn_pass(LayerPtrs p) {
  const int lane = (int)(threadIdx.x & 31u);
  const int half = lane >> 4;
  const int row = lane & 15;
  const int wslot = (int)(threadIdx.x >> 5);
  const int wave = (int)blockIdx.x * 8 + wslot;
  const int nwave = (int)gridDim.x * 8;
  const float invE = 1.0f / (float)N_EDGES_C;

  __shared__ f16_t smem[8][16 * 32];  // 8KB: one 16x32 f16 tile per wave
  f16_t* t = smem[wslot];

  v16h bW1e0 = vz16(), bW1e1 = vz16(), bW2e0 = vz16(), bW2e1 = vz16();
  v16h bW3e = vz16(), bW1v0 = vz16(), bW1v1 = vz16();
  v16h bW2v0 = vz16(), bW2v1 = vz16(), bW3v = vz16();

  if constexpr (STAGE <= 2) {
    bW1e0 = make_bfrag(p.pe + PE_W1, 9, 20, 0, lane);
    bW1e1 = make_bfrag(p.pe + PE_W1, 9, 20, 1, lane);
  }
  if constexpr (STAGE == 1 || STAGE == 2) {
    bW2e0 = make_bfrag(p.pe + PE_W2, 20, 20, 0, lane);
    bW2e1 = make_bfrag(p.pe + PE_W2, 20, 20, 1, lane);
  }
  if constexpr (STAGE == 2) bW3e = make_bfrag(p.pe + PE_W3, 20, 3, 0, lane);
  if constexpr (STAGE >= 2) {
    bW1v0 = make_bfrag(p.pv + PV_W1, 6, 20, 0, lane);
    bW1v1 = make_bfrag(p.pv + PV_W1, 6, 20, 1, lane);
  }
  if constexpr (STAGE >= 3) {
    bW2v0 = make_bfrag(p.pv + PV_W2, 20, 20, 0, lane);
    bW2v1 = make_bfrag(p.pv + PV_W2, 20, 20, 1, lane);
  }
  if constexpr (STAGE == 4) bW3v = make_bfrag(p.pv + PV_W3, 20, 3, 0, lane);

  float scE1[2], shE1[2], scE2[2], shE2[2], scV1[2], shV1[2], scV2[2], shV2[2];
  if constexpr (STAGE == 1 || STAGE == 2)
    bn_coeffs(p.st + 0, p.pe + PE_G1, p.pe + PE_BE1, 20, lane, invE, scE1, shE1);
  if constexpr (STAGE == 2)
    bn_coeffs(p.st + 64, p.pe + PE_G2, p.pe + PE_BE2, 20, lane, invE, scE2, shE2);
  if constexpr (STAGE >= 3)
    bn_coeffs(p.st + 128, p.pv + PV_G1, p.pv + PV_BE1, 20, lane, invE, scV1, shV1);
  if constexpr (STAGE == 4)
    bn_coeffs(p.st + 192, p.pv + PV_G2, p.pv + PV_BE2, 20, lane, invE, scV2, shV2);

  float aS[2] = {0.f, 0.f}, aQ[2] = {0.f, 0.f};

  for (int ti = wave; ti < NTILES_C; ti += nwave) {
    const int ebase = ti * 16;
    const int e0 = ebase + row;

    if constexpr (STAGE <= 2) {
      if (e0 + nwave * 16 < N_EDGES_C)  // -> global_prefetch_b8, next tile of H
        __builtin_prefetch(p.Hin + (size_t)(e0 + nwave * 16) * 3, 0, 1);

      // X1 = cat[Mi,Mj,H] @ W1e  (l1 bias cancelled by BN)
      v16h a0 = afrag_edge(p.Min, p.Hin, p.src, p.dst, e0, half);
      v8f d0[2] = {vz8(), vz8()};
      d0[0] = wmma32(a0, bW1e0, d0[0]);
      d0[1] = wmma32(a0, bW1e1, d0[1]);

      if constexpr (STAGE == 0) {
        acc_stats(d0[0], aS[0], aQ[0]);
        acc_stats(d0[1], aS[1], aQ[1]);
      } else {
        stage_act(t, d0, scE1, shE1, lane);
        __asm__ volatile("s_wait_dscnt 0" ::: "memory");
        v16h a1 = afrag_lds(t, lane);
        v8f d1[2] = {vz8(), vz8()};
        d1[0] = wmma32(a1, bW2e0, d1[0]);
        d1[1] = wmma32(a1, bW2e1, d1[1]);

        if constexpr (STAGE == 1) {
          acc_stats(d1[0], aS[0], aQ[0]);
          acc_stats(d1[1], aS[1], aQ[1]);
        } else {  // STAGE == 2: Hn out + first MLP_v layer stats
          stage_act(t, d1, scE2, shE2, lane);
          __asm__ volatile("s_wait_dscnt 0" ::: "memory");
          v16h a2 = afrag_lds(t, lane);
          v8f d2 = vz8();
          d2 = wmma32(a2, bW3e, d2);
          if (row < 3) {  // channels 0..2 carry Hn
            const float b3 = p.pe[PE_B3 + row];
#pragma unroll
            for (int r = 0; r < 8; ++r) {
              const int e = ebase + r + half * 8;
              const float v = d2[r] + b3;
              p.Hout[e * 3 + row] = v;
              t[(r + half * 8) * 32 + row] = (f16_t)v;  // stage for transpose
            }
          }
          __asm__ volatile("s_wait_dscnt 0" ::: "memory");
          v16h a3 = vz16();  // cat[Mi, Hn], K=6
          if (half == 0) {
            const int di = p.dst[e0] * 3;
            a3[0] = (f16_t)p.Min[di];
            a3[1] = (f16_t)p.Min[di + 1];
            a3[2] = (f16_t)p.Min[di + 2];
            a3[3] = t[row * 32 + 0];
            a3[4] = t[row * 32 + 1];
            a3[5] = t[row * 32 + 2];
          }
          v8f d3[2] = {vz8(), vz8()};
          d3[0] = wmma32(a3, bW1v0, d3[0]);
          d3[1] = wmma32(a3, bW1v1, d3[1]);
          acc_stats(d3[0], aS[0], aQ[0]);
          acc_stats(d3[1], aS[1], aQ[1]);
        }
      }
    } else {  // STAGE 3/4: restart MLP_v chain from persisted Hn
      if (e0 + nwave * 16 < N_EDGES_C)
        __builtin_prefetch(p.Hout + (size_t)(e0 + nwave * 16) * 3, 0, 1);

      v16h a3 = afrag_mihn(p.Min, p.Hout, p.dst, e0, half);
      v8f d3[2] = {vz8(), vz8()};
      d3[0] = wmma32(a3, bW1v0, d3[0]);
      d3[1] = wmma32(a3, bW1v1, d3[1]);
      stage_act(t, d3, scV1, shV1, lane);
      __asm__ volatile("s_wait_dscnt 0" ::: "memory");
      v16h a4 = afrag_lds(t, lane);
      v8f d4[2] = {vz8(), vz8()};
      d4[0] = wmma32(a4, bW2v0, d4[0]);
      d4[1] = wmma32(a4, bW2v1, d4[1]);

      if constexpr (STAGE == 3) {
        acc_stats(d4[0], aS[0], aQ[0]);
        acc_stats(d4[1], aS[1], aQ[1]);
      } else {  // STAGE == 4: messages + segment_sum via L2 atomics
        stage_act(t, d4, scV2, shV2, lane);
        __asm__ volatile("s_wait_dscnt 0" ::: "memory");
        v16h a5 = afrag_lds(t, lane);
        v8f d5 = vz8();
        d5 = wmma32(a5, bW3v, d5);
        if (row < 3) {
          const float b3 = p.pv[PV_B3 + row];
#pragma unroll
          for (int r = 0; r < 8; ++r) {
            const int e = ebase + r + half * 8;
            atomicAdd(&p.Mout[p.dst[e] * 3 + row], d5[r] + b3);
          }
        }
      }
    }
  }

  if constexpr (STAGE == 0) flush_stats(p.st + 0, aS, aQ, lane, 20);
  if constexpr (STAGE == 1) flush_stats(p.st + 64, aS, aQ, lane, 20);
  if constexpr (STAGE == 2) flush_stats(p.st + 128, aS, aQ, lane, 20);
  if constexpr (STAGE == 3) flush_stats(p.st + 192, aS, aQ, lane, 20);
}

__global__ void zero_layer(float* __restrict__ Mout, float* __restrict__ st) {
  const int i = (int)(blockIdx.x * blockDim.x + threadIdx.x);
  if (i < N_NODES_C * 3) Mout[i] = 0.f;
  if (i < 256) st[i] = 0.f;
}

__global__ void zero_buf(float* __restrict__ p, int n) {
  const int i = (int)(blockIdx.x * blockDim.x + threadIdx.x);
  if (i < n) p[i] = 0.f;
}

// Final 3->3->2 MLP (BN stats over E): scalar VALU, 3 passes.
// st layout: sum[0..2] @ [0..2], sumsq @ [4..6] (8 floats per BN).
template <int S>
__global__ __launch_bounds__(256) void out_mlp(const float* __restrict__ H,
                                               const float* __restrict__ po,
                                               float* st1, float* st2,
                                               float* __restrict__ out) {
  const float invE = 1.0f / (float)N_EDGES_C;
  float w1[9], w2[9], w3[6], b3[2];
#pragma unroll
  for (int i = 0; i < 9; ++i) w1[i] = po[PO_W1 + i];
#pragma unroll
  for (int i = 0; i < 9; ++i) w2[i] = po[PO_W2 + i];
#pragma unroll
  for (int i = 0; i < 6; ++i) w3[i] = po[PO_W3 + i];
  b3[0] = po[PO_B3];
  b3[1] = po[PO_B3 + 1];

  float sc1[3] = {0, 0, 0}, sh1[3] = {0, 0, 0}, sc2[3] = {0, 0, 0}, sh2[3] = {0, 0, 0};
  if constexpr (S >= 1) {
#pragma unroll
    for (int c = 0; c < 3; ++c) {
      const float mu = st1[c] * invE;
      const float var = st1[4 + c] * invE - mu * mu;
      const float rs = rsqrtf(var + 1e-5f);
      sc1[c] = po[PO_G1 + c] * rs;
      sh1[c] = po[PO_BE1 + c] - mu * sc1[c];
    }
  }
  if constexpr (S == 2) {
#pragma unroll
    for (int c = 0; c < 3; ++c) {
      const float mu = st2[c] * invE;
      const float var = st2[4 + c] * invE - mu * mu;
      const float rs = rsqrtf(var + 1e-5f);
      sc2[c] = po[PO_G2 + c] * rs;
      sh2[c] = po[PO_BE2 + c] - mu * sc2[c];
    }
  }

  float aS[3] = {0, 0, 0}, aQ[3] = {0, 0, 0};
  const int tid = (int)(blockIdx.x * blockDim.x + threadIdx.x);
  const int nth = (int)(gridDim.x * blockDim.x);
  for (int e = tid; e < N_EDGES_C; e += nth) {
    const float h0 = H[3 * e], h1 = H[3 * e + 1], h2 = H[3 * e + 2];
    float z[3];
#pragma unroll
    for (int n = 0; n < 3; ++n) z[n] = h0 * w1[n] + h1 * w1[3 + n] + h2 * w1[6 + n];
    if constexpr (S == 0) {
#pragma unroll
      for (int n = 0; n < 3; ++n) { aS[n] += z[n]; aQ[n] += z[n] * z[n]; }
    } else {
      float zr[3];
#pragma unroll
      for (int n = 0; n < 3; ++n) zr[n] = fmaxf(z[n] * sc1[n] + sh1[n], 0.f);
      float y[3];
#pragma unroll
      for (int n = 0; n < 3; ++n)
        y[n] = zr[0] * w2[n] + zr[1] * w2[3 + n] + zr[2] * w2[6 + n];
      if constexpr (S == 1) {
#pragma unroll
        for (int n = 0; n < 3; ++n) { aS[n] += y[n]; aQ[n] += y[n] * y[n]; }
      } else {
        float yr[3];
#pragma unroll
        for (int n = 0; n < 3; ++n) yr[n] = fmaxf(y[n] * sc2[n] + sh2[n], 0.f);
        out[2 * e]     = yr[0] * w3[0] + yr[1] * w3[2] + yr[2] * w3[4] + b3[0];
        out[2 * e + 1] = yr[0] * w3[1] + yr[1] * w3[3] + yr[2] * w3[5] + b3[1];
      }
    }
  }

  if constexpr (S < 2) {
    __shared__ float red[6];
    if (threadIdx.x < 6) red[threadIdx.x] = 0.f;
    __syncthreads();
#pragma unroll
    for (int n = 0; n < 3; ++n) {
      atomicAdd(&red[n], aS[n]);      // ds_add_f32
      atomicAdd(&red[3 + n], aQ[n]);
    }
    __syncthreads();
    float* tgt = (S == 0) ? st1 : st2;
    if (threadIdx.x < 3) {
      atomicAdd(&tgt[threadIdx.x], red[threadIdx.x]);
      atomicAdd(&tgt[4 + threadIdx.x], red[3 + threadIdx.x]);
    }
  }
}

extern "C" void kernel_launch(void* const* d_in, const int* in_sizes, int n_in,
                              void* d_out, int out_size, void* d_ws, size_t ws_size,
                              hipStream_t stream) {
  (void)in_sizes; (void)n_in; (void)out_size; (void)ws_size;
  const float* M0 = (const float*)d_in[0];   // [100000,3]
  const float* H0 = (const float*)d_in[1];   // [3200000,3]
  const int* ei   = (const int*)d_in[2];     // [2,3200000]
  const float* pe = (const float*)d_in[3];   // 763 floats
  const float* pv = (const float*)d_in[4];   // 703 floats
  const float* po = (const float*)d_in[5];   // 44 floats
  const int* src = ei;                       // edge_index[0]
  const int* dst = ei + N_EDGES_C;           // edge_index[1]

  // Workspace: 2x M (0.6M f) + 2x H (19.2M f) + stats (~272 f) ~= 79.2 MB.
  float* w = (float*)d_ws;
  float* Mb[2] = {w, w + (size_t)N_NODES_C * 3};
  float* Hb[2] = {w + (size_t)N_NODES_C * 6,
                  w + (size_t)N_NODES_C * 6 + (size_t)N_EDGES_C * 3};
  float* st = w + (size_t)N_NODES_C * 6 + (size_t)N_EDGES_C * 6;  // 256 floats
  float* ost = st + 256;                                          // 16 floats

  const dim3 blk(256);     // 8 wave32 waves / block
  const int gridP = 2000;  // 16000 waves, ~12.5 tiles each (grid-stride)

  for (int l = 0; l < NUM_LAYER_C; ++l) {
    LayerPtrs p;
    p.Min = (l == 0) ? M0 : Mb[(l - 1) & 1];
    p.Hin = (l == 0) ? H0 : Hb[(l - 1) & 1];
    p.src = src;
    p.dst = dst;
    p.Hout = Hb[l & 1];
    p.Mout = Mb[l & 1];
    p.pe = pe;
    p.pv = pv;
    p.st = st;
    zero_layer<<<(N_NODES_C * 3 + 255) / 256, blk, 0, stream>>>(p.Mout, st);
    mpn_pass<0><<<gridP, blk, 0, stream>>>(p);
    mpn_pass<1><<<gridP, blk, 0, stream>>>(p);
    mpn_pass<2><<<gridP, blk, 0, stream>>>(p);
    mpn_pass<3><<<gridP, blk, 0, stream>>>(p);
    mpn_pass<4><<<gridP, blk, 0, stream>>>(p);
  }

  zero_buf<<<1, 64, 0, stream>>>(ost, 16);
  const float* Hf = Hb[(NUM_LAYER_C - 1) & 1];
  out_mlp<0><<<2048, blk, 0, stream>>>(Hf, po, ost, ost + 8, (float*)d_out);
  out_mlp<1><<<2048, blk, 0, stream>>>(Hf, po, ost, ost + 8, (float*)d_out);
  out_mlp<2><<<2048, blk, 0, stream>>>(Hf, po, ost, ost + 8, (float*)d_out);
}
